// Model_76227079570082
// MI455X (gfx1250) — matrix operand order repaired and checked by reading, not compiled, not run on hardware
//
#include <hip/hip_runtime.h>

// ---------------------------------------------------------------------------
// CDNA5 (gfx1250) sparse-UNet.
// Hot path: bf16 WMMA gather-GEMM (v_wmma_f32_16x16x32_bf16), f32 accumulate.
// Activations are pre-activated (fused BN+ReLU) into an (N+1)-row buffer with
// a zero sentinel row, so the gather inner loop has NO guards: 4x b128 loads
// for A, 4x b128 loads for pre-packed bf16 B, 2 independent WMMAs per chunk.
// ---------------------------------------------------------------------------

typedef __attribute__((ext_vector_type(16))) __bf16       v16bf;
typedef __attribute__((ext_vector_type(8)))  float        v8f;
typedef __attribute__((ext_vector_type(4)))  float        v4f;
typedef __attribute__((ext_vector_type(4)))  unsigned int v4u;

#define BN_EPS 1e-4f

// ------------------------------- kernels -----------------------------------

__global__ void k_zero(float* __restrict__ p, long n) {
  long i = (long)blockIdx.x * blockDim.x + threadIdx.x;
  if (i < n) p[i] = 0.f;
}

__global__ void k_add(float* __restrict__ x, const float* __restrict__ t, long n) {
  long i = (long)blockIdx.x * blockDim.x + threadIdx.x;
  if (i < n) x[i] += t[i];
}

// dst[n*dstStride + c] = src[n*C + c]
__global__ void k_copy_cols(float* __restrict__ dst, int dstStride,
                            const float* __restrict__ src, int N, int C) {
  long i = (long)blockIdx.x * blockDim.x + threadIdx.x;
  if (i >= (long)N * C) return;
  int n = (int)(i / C), c = (int)(i % C);
  dst[(size_t)n * dstStride + c] = src[i];
}

// Per-channel BN stats -> fused scale/shift:  y = x*ss[c] + ss[C+c]
__global__ __launch_bounds__(256) void k_bn(const float* __restrict__ f,
                                            const float* __restrict__ g,
                                            const float* __restrict__ b,
                                            float* __restrict__ ss, int N, int C) {
  int c = blockIdx.x;
  float s = 0.f, s2 = 0.f;
  for (int n = threadIdx.x; n < N; n += 256) {
    float v = f[(size_t)n * C + c];
    s += v; s2 += v * v;
  }
  __shared__ float sh[256], sh2[256];
  sh[threadIdx.x] = s; sh2[threadIdx.x] = s2;
  __syncthreads();
  for (int st = 128; st > 0; st >>= 1) {
    if (threadIdx.x < st) {
      sh[threadIdx.x]  += sh[threadIdx.x + st];
      sh2[threadIdx.x] += sh2[threadIdx.x + st];
    }
    __syncthreads();
  }
  if (threadIdx.x == 0) {
    float inv = 1.f / (float)N;
    float mu  = sh[0] * inv;
    float var = sh2[0] * inv - mu * mu;
    float rs  = rsqrtf(var + BN_EPS);
    float sc  = g[c] * rs;
    ss[c]     = sc;
    ss[C + c] = b[c] - mu * sc;
  }
}

// act[(N+1) x C]: rows <N get relu(x*ss[c]+ss[C+c]); row N (sentinel) = 0
__global__ void k_bnapply(float* __restrict__ act, const float* __restrict__ x,
                          const float* __restrict__ ss, int N, int C) {
  long i = (long)blockIdx.x * blockDim.x + threadIdx.x;
  if (i >= (long)(N + 1) * C) return;
  int n = (int)(i / C), c = (int)(i % C);
  act[i] = (n < N) ? fmaxf(fmaf(x[i], ss[c], ss[C + c]), 0.f) : 0.f;
}

// pad colors (N x 3) -> act ((N+1) x 32), zeros elsewhere (raw, no BN)
__global__ void k_pad_in(float* __restrict__ act, const float* __restrict__ col, int N) {
  long i = (long)blockIdx.x * blockDim.x + threadIdx.x;
  if (i >= (long)(N + 1) * 32) return;
  int n = (int)(i >> 5), c = (int)(i & 31);
  act[i] = (n < N && c < 3) ? col[n * 3 + c] : 0.f;
}

// Pack W (K,Cs,D) f32 -> bf16 in B-fragment lane order.
// Layout: [(k*nch + cc)*ndt + dt][lane][16 bf16], lane's 16 values contiguous.
// Element pair t: c = cc*32 + 8*(lane>>4) + 2*(dword&3) + 16*(dword>>2), d = dt*16 + (lane&15).
__global__ void k_packw(__bf16* __restrict__ wp, const float* __restrict__ W,
                        int K, int Cs, int C, int D, long totalPairs) {
  long t = (long)blockIdx.x * blockDim.x + threadIdx.x;
  if (t >= totalPairs) return;
  int dword = (int)(t & 7);
  long r = t >> 3;
  int lane = (int)(r & 31); r >>= 5;
  int ndt = D >> 4;
  int dt = (int)(r % ndt); r /= ndt;
  int nch = C >> 5;
  int cc = (int)(r % nch);
  int k  = (int)(r / nch);
  int c = (cc << 5) + ((lane >> 4) << 3) + 2 * (dword & 3) + ((dword >> 2) << 4);
  int d = (dt << 4) + (lane & 15);
  float w0 = (c     < Cs) ? W[((size_t)k * Cs + c)     * D + d] : 0.f;
  float w1 = (c + 1 < Cs) ? W[((size_t)k * Cs + c + 1) * D + d] : 0.f;
  wp[2 * t]     = (__bf16)w0;
  wp[2 * t + 1] = (__bf16)w1;
}

// Gather-GEMM: out[n,d] = sum_k sum_c fp[idx(n,k), c] * W[k,c,d]
//   fp has (rows+1) x C with zero sentinel row when nbr != null.
//   C % 32 == 0, D % 32 == 0. One wave -> 16x32 output tile. EXEC all-1s.
__global__ __launch_bounds__(32) void k_gg(float* __restrict__ out,
                                           const float* __restrict__ fp,
                                           const int*   __restrict__ nbr,
                                           const __bf16* __restrict__ wp,
                                           int N, int C, int D, int K) {
  const int n0   = blockIdx.x << 4;
  const int dt0  = blockIdx.y << 1;      // two 16-wide D tiles
  const int lane = threadIdx.x;
  const int m    = lane & 15;
  const int half = lane >> 4;
  int row = n0 + m; if (row >= N) row = N - 1;   // clamp (store guarded)
  const int nch = C >> 5;
  const int ndt = D >> 4;

  v8f acc0 = {}, acc1 = {};
  for (int k = 0; k < K; ++k) {
    const int idx = nbr ? nbr[(size_t)row * K + k] : row;
    const float* frow = fp + (size_t)idx * C;
    const __bf16* wk  = wp + ((size_t)k * nch * ndt + (size_t)dt0) * 512;
    for (int cc = 0; cc < nch; ++cc) {
      // ---- A fragment: two contiguous 8-float runs per lane ----
      const float* pa = frow + (cc << 5) + (half << 3);
      v4f a0 = *(const v4f*)(pa);
      v4f a1 = *(const v4f*)(pa + 4);
      v4f a2 = *(const v4f*)(pa + 16);
      v4f a3 = *(const v4f*)(pa + 20);
      v16bf a;
#pragma unroll
      for (int j = 0; j < 4; ++j) {
        a[j]      = (__bf16)a0[j];
        a[4 + j]  = (__bf16)a1[j];
        a[8 + j]  = (__bf16)a2[j];
        a[12 + j] = (__bf16)a3[j];
      }
      // ---- B fragments: pre-packed, 16 bf16 contiguous per lane ----
      const __bf16* pb = wk + (size_t)cc * ndt * 512 + lane * 16;
      union { v4u u[2]; v16bf b; } cb0, cb1;
      cb0.u[0] = *(const v4u*)(pb);
      cb0.u[1] = *(const v4u*)(pb + 8);
      cb1.u[0] = *(const v4u*)(pb + 512);
      cb1.u[1] = *(const v4u*)(pb + 520);
      acc0 = __builtin_amdgcn_wmma_f32_16x16x32_bf16(
          false, a, false, cb0.b, (short)0, acc0, false, false);
      acc1 = __builtin_amdgcn_wmma_f32_16x16x32_bf16(
          false, a, false, cb1.b, (short)0, acc1, false, false);
    }
  }
  const int dbase = dt0 << 4;
#pragma unroll
  for (int r = 0; r < 8; ++r) {
    int rn = n0 + r + 8 * half;
    if (rn < N) {
      out[(size_t)rn * D + dbase + m]      = acc0[r];
      out[(size_t)rn * D + dbase + 16 + m] = acc1[r];
    }
  }
}

// Deconv scatter: out[pool[m,k], outOff+d] += sum_c bnrelu(f[m,c]) * W[k,c,d]
__global__ __launch_bounds__(128) void k_up(float* __restrict__ out, int outStride,
                                            int outOff,
                                            const float* __restrict__ f,
                                            const int*   __restrict__ pool,
                                            const float* __restrict__ W,
                                            const float* __restrict__ ss,
                                            int Nc, int Nfine, int C, int D) {
  int m = blockIdx.x;
  extern __shared__ float sf[];
  for (int c = threadIdx.x; c < C; c += blockDim.x) {
    float v = f[(size_t)m * C + c];
    sf[c] = fmaxf(fmaf(v, ss[c], ss[C + c]), 0.f);
  }
  __syncthreads();
  for (int k = 0; k < 8; ++k) {
    int idx = pool[(size_t)m * 8 + k];
    if (idx < 0 || idx >= Nfine) continue;
    for (int d = threadIdx.x; d < D; d += blockDim.x) {
      const float* wpt = W + (size_t)k * C * D + d;
      float acc = 0.f;
      for (int c = 0; c < C; ++c) acc = fmaf(sf[c], wpt[(size_t)c * D], acc);
      atomicAdd(&out[(size_t)idx * outStride + outOff + d], acc);
    }
  }
}

// Final BN+ReLU + two linear heads.  out = [sem (N*20) | nb (N*6)] flat.
__global__ __launch_bounds__(32) void k_heads(float* __restrict__ out,
                                              const float* __restrict__ x,
                                              const float* __restrict__ ss,
                                              const float* __restrict__ lw,
                                              const float* __restrict__ lb,
                                              const float* __restrict__ nw,
                                              const float* __restrict__ nbb,
                                              int N) {
  int n = blockIdx.x, t = threadIdx.x;
  __shared__ float a[32];
  float v = x[(size_t)n * 32 + t];
  a[t] = fmaxf(fmaf(v, ss[t], ss[32 + t]), 0.f);
  __syncthreads();
  if (t < 20) {
    float acc = lb[t];
    for (int c = 0; c < 32; ++c) acc = fmaf(a[c], lw[c * 20 + t], acc);
    out[(size_t)n * 20 + t] = acc;
  } else if (t < 26) {
    int j = t - 20;
    float acc = nbb[j];
    for (int c = 0; c < 32; ++c) acc = fmaf(a[c], nw[c * 6 + j], acc);
    out[(size_t)N * 20 + (size_t)n * 6 + j] = acc;
  }
}

// ------------------------------ host side ----------------------------------

namespace {

const int NP_[7] = {32, 64, 96, 128, 160, 192, 224};
const int NS_[7] = {50000, 25000, 12500, 6250, 3125, 1562, 781};

struct Ctx {
  hipStream_t s;
  void* const* din;
  int pc;
  char*  ws;
  size_t off, cap;
  float* bnbuf;
  const int* nbrs[7];
  const int* pools[6];
};

inline const float* nextp(Ctx& c) { return (const float*)c.din[c.pc++]; }

inline void* bump_bytes(Ctx& c, size_t bytes) {
  bytes = (bytes + 255) & ~(size_t)255;
  void* p = (void*)(c.ws + c.off);
  c.off += bytes;
  return p;
}
inline float* bump(Ctx& c, size_t elems) {
  return (float*)bump_bytes(c, elems * sizeof(float));
}

inline void run_zero(Ctx& c, float* p, long n) {
  k_zero<<<(int)((n + 255) / 256), 256, 0, c.s>>>(p, n);
}
inline void run_add(Ctx& c, float* x, const float* t, long n) {
  k_add<<<(int)((n + 255) / 256), 256, 0, c.s>>>(x, t, n);
}
inline void run_copy(Ctx& c, float* dst, int dstStride, const float* src, int N, int C) {
  long n = (long)N * C;
  k_copy_cols<<<(int)((n + 255) / 256), 256, 0, c.s>>>(dst, dstStride, src, N, C);
}
inline void run_bn(Ctx& c, const float* f, const float* g, const float* b, int N, int C) {
  k_bn<<<C, 256, 0, c.s>>>(f, g, b, c.bnbuf, N, C);
}
inline void run_bnapply(Ctx& c, float* act, const float* x, int N, int C) {
  long n = (long)(N + 1) * C;
  k_bnapply<<<(int)((n + 255) / 256), 256, 0, c.s>>>(act, x, c.bnbuf, N, C);
}

// pack W (K,Cs,D) to bf16 fragments (padded to C channels), run gather-GEMM
inline void run_gg(Ctx& c, float* out, const float* fp, const int* nbr,
                   const float* W, int Cs, int N, int C, int D, int K) {
  size_t mk = c.off;
  long totalPairs = (long)K * (C >> 5) * (D >> 4) * 32 * 8;
  __bf16* wp = (__bf16*)bump_bytes(c, (size_t)totalPairs * 2 * sizeof(__bf16));
  k_packw<<<(int)((totalPairs + 255) / 256), 256, 0, c.s>>>(wp, W, K, Cs, C, D,
                                                            totalPairs);
  dim3 g((N + 15) / 16, D / 32);
  k_gg<<<g, 32, 0, c.s>>>(out, fp, nbr, wp, N, C, D, K);
  c.off = mk;   // safe: stream-ordered reuse
}
inline void run_up(Ctx& c, float* out, int outStride, int outOff, const float* f,
                   const int* pool, const float* W, int Nc, int Nfine, int C, int D) {
  k_up<<<Nc, 128, C * sizeof(float), c.s>>>(out, outStride, outOff, f, pool, W,
                                            c.bnbuf, Nc, Nfine, C, D);
}

// residual block (a == b), in place on x
void run_block_same(Ctx& cx, float* x, const int* nbr, int N, int C,
                    const float* g1, const float* b1, const float* W1,
                    const float* g2, const float* b2, const float* W2) {
  size_t mk = cx.off;
  float* act = bump(cx, (size_t)(N + 1) * C);
  float* t1  = bump(cx, (size_t)N * C);
  float* t2  = bump(cx, (size_t)N * C);
  run_bn(cx, x, g1, b1, N, C);
  run_bnapply(cx, act, x, N, C);
  run_gg(cx, t1, act, nbr, W1, C, N, C, C, 27);
  run_bn(cx, t1, g2, b2, N, C);
  run_bnapply(cx, act, t1, N, C);
  run_gg(cx, t2, act, nbr, W2, C, N, C, C, 27);
  run_add(cx, x, t2, (long)N * C);
  cx.off = mk;
}

float* unet_run(Ctx& cx, int lvl, float* x) {
  const int N = NS_[lvl], C = NP_[lvl];
  const int* nbr = cx.nbrs[lvl];

  // ---- encoder blocks (2x, in place) ----
  for (int bi = 0; bi < 2; ++bi) {
    const float* g1 = nextp(cx); const float* b1 = nextp(cx); const float* W1 = nextp(cx);
    const float* g2 = nextp(cx); const float* b2 = nextp(cx); const float* W2 = nextp(cx);
    run_block_same(cx, x, nbr, N, C, g1, b1, W1, g2, b2, W2);
  }
  if (lvl == 6) return x;

  // ---- params in flatten order: dec blocks first, then down/up ----
  const float* d1g1 = nextp(cx); const float* d1b1 = nextp(cx); const float* d1W1 = nextp(cx);
  const float* d1g2 = nextp(cx); const float* d1b2 = nextp(cx); const float* d1W2 = nextp(cx);
  const float* d1nin = nextp(cx);
  const float* d2g1 = nextp(cx); const float* d2b1 = nextp(cx); const float* d2W1 = nextp(cx);
  const float* d2g2 = nextp(cx); const float* d2b2 = nextp(cx); const float* d2W2 = nextp(cx);
  const float* dng = nextp(cx); const float* dnb = nextp(cx); const float* dnW = nextp(cx);
  const float* upg = nextp(cx); const float* upb = nextp(cx); const float* upW = nextp(cx);

  const int N1 = NS_[lvl + 1], C1 = NP_[lvl + 1];
  size_t mk = cx.off;
  float* cat = bump(cx, (size_t)N * 2 * C);
  float* d   = bump(cx, (size_t)N1 * C1);

  // down: strided conv (K=8 children), fused BN+ReLU on fine features
  {
    size_t mk2 = cx.off;
    float* act = bump(cx, (size_t)(N + 1) * C);
    run_bn(cx, x, dng, dnb, N, C);
    run_bnapply(cx, act, x, N, C);
    run_gg(cx, d, act, cx.pools[lvl], dnW, C, N1, C, C1, 8);
    cx.off = mk2;
  }

  // save skip into concat[:, :C]; zero the up half
  run_zero(cx, cat, (long)N * 2 * C);
  run_copy(cx, cat, 2 * C, x, N, C);

  d = unet_run(cx, lvl + 1, d);

  // up: deconv scatter-add into concat[:, C:2C]
  run_bn(cx, d, upg, upb, N1, C1);
  run_up(cx, cat, 2 * C, C, d, cx.pools[lvl], upW, N1, N, C1, C);

  // ---- decoder block 1: (2C -> C), output into x (skip content now dead) ----
  {
    size_t mk2 = cx.off;
    float* act = bump(cx, (size_t)(N + 1) * 2 * C);
    float* t1  = bump(cx, (size_t)N * C);
    run_bn(cx, cat, d1g1, d1b1, N, 2 * C);
    run_bnapply(cx, act, cat, N, 2 * C);
    run_gg(cx, t1, act, nbr, d1W1, 2 * C, N, 2 * C, C, 27);
    run_bn(cx, t1, d1g2, d1b2, N, C);
    run_bnapply(cx, act, t1, N, C);
    run_gg(cx, x, act, nbr, d1W2, C, N, C, C, 27);
    // identity path: cat @ nin (raw input, no BN, no sentinel access) -> t1
    run_gg(cx, t1, cat, nullptr, d1nin, 2 * C, N, 2 * C, C, 1);
    run_add(cx, x, t1, (long)N * C);
    cx.off = mk2;
  }
  // ---- decoder block 2: (C -> C), in place ----
  run_block_same(cx, x, nbr, N, C, d2g1, d2b1, d2W1, d2g2, d2b2, d2W2);

  cx.off = mk;
  return x;
}

} // namespace

extern "C" void kernel_launch(void* const* d_in, const int* in_sizes, int n_in,
                              void* d_out, int out_size, void* d_ws, size_t ws_size,
                              hipStream_t stream) {
  (void)in_sizes; (void)n_in; (void)out_size;
  Ctx cx;
  cx.s = stream; cx.din = d_in;
  cx.ws = (char*)d_ws; cx.off = 0; cx.cap = ws_size;

  const float* colors = (const float*)d_in[0];
  for (int l = 0; l < 7; ++l) cx.nbrs[l]  = (const int*)d_in[1 + l];
  for (int l = 0; l < 6; ++l) cx.pools[l] = (const int*)d_in[8 + l];
  cx.pc = 14;

  cx.bnbuf = bump(cx, 1024);                 // fused BN scale/shift (max 2*384)
  const float* in_W = nextp(cx);             // (27, 3, 32)

  float* x0 = bump(cx, (size_t)NS_[0] * NP_[0]);
  {
    size_t mk = cx.off;
    float* actIn = bump(cx, (size_t)(NS_[0] + 1) * 32);
    long n = (long)(NS_[0] + 1) * 32;
    k_pad_in<<<(int)((n + 255) / 256), 256, 0, stream>>>(actIn, colors, NS_[0]);
    // input SubmanifoldConvolution 3->32 (zero-padded to 32), no BN/ReLU
    run_gg(cx, x0, actIn, cx.nbrs[0], in_W, 3, NS_[0], 32, 32, 27);
    cx.off = mk;
  }

  float* xr = unet_run(cx, 0, x0);

  const float* fg = nextp(cx); const float* fb = nextp(cx);
  const float* lw = nextp(cx); const float* lb = nextp(cx);
  const float* nw = nextp(cx); const float* nbb = nextp(cx);

  k_bn<<<32, 256, 0, stream>>>(xr, fg, fb, cx.bnbuf, NS_[0], 32);
  k_heads<<<NS_[0], 32, 0, stream>>>((float*)d_out, xr, cx.bnbuf, lw, lb, nw, nbb, NS_[0]);
}
